// OuterProducterMean_89713276879142
// MI455X (gfx1250) — matrix operand order repaired and checked by reading, not compiled, not run on hardware
//
#include <hip/hip_runtime.h>
#include <hip/hip_bf16.h>

// Problem constants (N=1)
#define S_DIM 128
#define R_DIM 256
#define CM_DIM 256
#define C_DIM 32
#define CZ_DIM 128

typedef __attribute__((ext_vector_type(16))) __bf16 v16bf;
typedef __attribute__((ext_vector_type(2)))  __bf16 v2bf;
typedef __attribute__((ext_vector_type(8)))  float  v8f;
typedef __attribute__((ext_vector_type(8)))  unsigned int v8u;

// ---- helpers -------------------------------------------------------------

static __device__ inline unsigned short f2bf(float f) {
    union { float f; unsigned u; } v; v.f = f;
    unsigned u = v.u;
    unsigned r = u + 0x7FFFu + ((u >> 16) & 1u);   // round-to-nearest-even
    return (unsigned short)(r >> 16);
}

static __device__ inline unsigned pack_bf16x2(float lo, float hi) {
#if __has_builtin(__builtin_amdgcn_cvt_pk_bf16_f32)
    v2bf p = __builtin_amdgcn_cvt_pk_bf16_f32(lo, hi);
    return __builtin_bit_cast(unsigned, p);
#else
    return (unsigned)f2bf(lo) | ((unsigned)f2bf(hi) << 16);
#endif
}

// A-fragment (16xK slice, M x 32 bf16), row-major rows of length rowStride.
// ISA layout: lane m = l&15, half h = l>>4.
// VGPR j(0..7): K = 16h + 2j, +1 (contiguous k used here, not ISA table order).
static __device__ inline v16bf load_frag_A(const unsigned short* base,
                                           int rowStride, int k0, int lane) {
    int m = lane & 15, h = lane >> 4;
    const unsigned short* p = base + (size_t)m * rowStride + k0;
    v8u u;
#pragma unroll
    for (int j = 0; j < 8; ++j)
        u[j] = *(const unsigned int*)(p + 16 * h + 2 * j);
    return __builtin_bit_cast(v16bf, u);
}

// B-fragment (K x 16 slice), stored column-major: column n contiguous in K.
// Contiguous k order used here (same as A): lane n = l&15; lanes 0-15 hold K=0..15, lanes 16-31 K=16..31.
static __device__ inline v16bf load_frag_B(const unsigned short* base,
                                           int colStride, int k0, int lane) {
    int n = lane & 15, h = lane >> 4;
    const unsigned short* p = base + (size_t)n * colStride + k0 + 16 * h;
    v8u u;
#pragma unroll
    for (int j = 0; j < 8; ++j)
        u[j] = *(const unsigned int*)(p + 2 * j);
    return __builtin_bit_cast(v16bf, u);
}

// ---- kernel 1: projections a = msa@left, b = msa@right -------------------
// Writes a_ws[(r*32+c)*128 + s], b_ws[(t*32+e)*128 + s] as bf16
// (A-fragment-friendly for 'a', B-fragment-friendly for 'b').
__global__ __launch_bounds__(256)
void opm_proj_kernel(const float* __restrict__ msa,
                     const float* __restrict__ left,
                     const float* __restrict__ right,
                     unsigned short* __restrict__ a_ws,
                     unsigned short* __restrict__ b_ws) {
    __shared__ float msaRow[4][CM_DIM];
    const int r   = blockIdx.x;
    const int tid = threadIdx.x;

    const int s_local = tid >> 6;          // 0..3
    const int which   = (tid >> 5) & 1;    // 0 = left/a, 1 = right/b
    const int c       = tid & 31;
    const float* proj = which ? right : left;
    unsigned short* dst = which ? b_ws : a_ws;

    for (int s0 = 0; s0 < S_DIM; s0 += 4) {
        for (int i = tid; i < 4 * CM_DIM; i += 256) {
            int row = i >> 8, col = i & (CM_DIM - 1);
            msaRow[row][col] = msa[((size_t)(s0 + row) * R_DIM + r) * CM_DIM + col];
        }
        __syncthreads();

        float acc = 0.f;
#pragma unroll 8
        for (int m = 0; m < CM_DIM; ++m)
            acc += msaRow[s_local][m] * proj[m * C_DIM + c];

        dst[((size_t)r * C_DIM + c) * S_DIM + (s0 + s_local)] = f2bf(acc);
        __syncthreads();
    }
}

// ---- kernel 2: out_proj -> Wt bf16 with permuted K -----------------------
// Phase-3 contraction index is permuted: k' = e*32 + c  (c = fast outer dim).
// Wt[z*1024 + k'] = W[(c*32 + e)*128 + z]
__global__ __launch_bounds__(256)
void opm_wt_kernel(const float* __restrict__ W, unsigned short* __restrict__ Wt) {
    int idx = blockIdx.x * 256 + threadIdx.x;          // 0 .. 131071
    int z = idx >> 10, kp = idx & 1023;
    int c = kp & 31, e = kp >> 5;
    Wt[idx] = f2bf(W[((size_t)c * C_DIM + e) * CZ_DIM + z]);
}

// ---- kernel 3: fused outer-product + output projection -------------------
// grid.x = t (256), grid.y = r-block (16). 256 threads = 8 waves.
#define STG_STRIDE 1058   // 529 dwords; 529 % 64 = 17 -> conflict-free rows

__global__ __launch_bounds__(256)
void opm_main_kernel(const unsigned short* __restrict__ a_ws,
                     const unsigned short* __restrict__ b_ws,
                     const unsigned short* __restrict__ Wt,
                     float* __restrict__ out) {
    __shared__ unsigned short staged[16][STG_STRIDE];  // 16 pairs x 1024 (padded)

    const int t    = blockIdx.x;
    const int r0   = blockIdx.y * 16;
    const int lane = threadIdx.x & 31;
    const int wave = threadIdx.x >> 5;                 // 0..7

    // ---- Phase A: outer[p][c][e] = sum_s a[r0+p,s,c] * b[t,s,e] ----------
    // 16 pairs x (2x2) 16x16 tiles = 64 tile jobs over 8 waves.
#pragma unroll 2
    for (int job = wave; job < 64; job += 8) {
        const int p  = job >> 2;
        const int ct = (job >> 1) & 1;
        const int et = job & 1;

        const unsigned short* abase =
            a_ws + ((size_t)(r0 + p) * C_DIM + ct * 16) * S_DIM;
        const unsigned short* bbase =
            b_ws + ((size_t)t * C_DIM + et * 16) * S_DIM;

        // Two independent accumulator chains (even/odd K-steps).
        v8f acc0 = {}, acc1 = {};
#pragma unroll
        for (int kk = 0; kk < 2; ++kk) {
            v16bf A0 = load_frag_A(abase, S_DIM, (2 * kk) * 32, lane);
            v16bf B0 = load_frag_B(bbase, S_DIM, (2 * kk) * 32, lane);
            acc0 = __builtin_amdgcn_wmma_f32_16x16x32_bf16(
                false, A0, false, B0, (short)0, acc0, false, false);
            v16bf A1 = load_frag_A(abase, S_DIM, (2 * kk + 1) * 32, lane);
            v16bf B1 = load_frag_B(bbase, S_DIM, (2 * kk + 1) * 32, lane);
            acc1 = __builtin_amdgcn_wmma_f32_16x16x32_bf16(
                false, A1, false, B1, (short)0, acc1, false, false);
        }
        v8f acc = acc0 + acc1;

        // D layout: n = lane&15 (e), m = 8*(lane>>4)+i (c).
        // staged index k' = e*32 + c -> 8 contiguous bf16 per lane.
        const int n = lane & 15, h = lane >> 4;
        const int e = et * 16 + n;
        unsigned* srow =
            (unsigned*)&staged[p][e * C_DIM + ct * 16 + 8 * h];
#pragma unroll
        for (int i = 0; i < 4; ++i)
            srow[i] = pack_bf16x2(acc[2 * i], acc[2 * i + 1]);
    }
    __syncthreads();

    // ---- Phase B: out[p][z] = sum_k' staged[p][k'] * Wt[z][k'] -----------
    // One 16-wide z-tile per wave; K = 1024 -> 32 wmma steps, 2 chains.
    const unsigned short* wbase = Wt + (size_t)wave * 16 * (C_DIM * C_DIM);
    v8f acc0 = {}, acc1 = {};
#pragma unroll 2
    for (int ks = 0; ks < 16; ++ks) {
        v16bf A0 = load_frag_A(&staged[0][0], STG_STRIDE, (2 * ks) * 32, lane);
        v16bf B0 = load_frag_B(wbase, C_DIM * C_DIM, (2 * ks) * 32, lane);
        acc0 = __builtin_amdgcn_wmma_f32_16x16x32_bf16(
            false, A0, false, B0, (short)0, acc0, false, false);
        v16bf A1 = load_frag_A(&staged[0][0], STG_STRIDE, (2 * ks + 1) * 32, lane);
        v16bf B1 = load_frag_B(wbase, C_DIM * C_DIM, (2 * ks + 1) * 32, lane);
        acc1 = __builtin_amdgcn_wmma_f32_16x16x32_bf16(
            false, A1, false, B1, (short)0, acc1, false, false);
    }
    v8f acc = acc0 + acc1;

    const int n = lane & 15, h = lane >> 4;
#pragma unroll
    for (int i = 0; i < 8; ++i) {
        int p = 8 * h + i;
        int z = wave * 16 + n;
        out[((size_t)(r0 + p) * R_DIM + t) * CZ_DIM + z] = acc[i];
    }
}

// ---- launch --------------------------------------------------------------
extern "C" void kernel_launch(void* const* d_in, const int* in_sizes, int n_in,
                              void* d_out, int out_size, void* d_ws, size_t ws_size,
                              hipStream_t stream) {
    const float* msa   = (const float*)d_in[0];   // [1,128,256,256]
    const float* left  = (const float*)d_in[1];   // [256,32]
    const float* right = (const float*)d_in[2];   // [256,32]
    const float* outp  = (const float*)d_in[3];   // [1024,128]
    float* out = (float*)d_out;                   // [1,256,256,128]

    unsigned short* ws   = (unsigned short*)d_ws;
    unsigned short* a_ws = ws;                                   // 2 MB
    unsigned short* b_ws = ws + (size_t)R_DIM * C_DIM * S_DIM;   // 2 MB
    unsigned short* Wt   = b_ws + (size_t)R_DIM * C_DIM * S_DIM; // 256 KB

    opm_proj_kernel<<<R_DIM, 256, 0, stream>>>(msa, left, right, a_ws, b_ws);
    opm_wt_kernel<<<(C_DIM * C_DIM * CZ_DIM) / 256, 256, 0, stream>>>(outp, Wt);
    opm_main_kernel<<<dim3(R_DIM, R_DIM / 16), 256, 0, stream>>>(a_ws, b_ws, Wt, out);
}
